// CustomLSTM_23510650978875
// MI455X (gfx1250) — matrix operand order repaired and checked by reading, not compiled, not run on hardware
//
#include <hip/hip_runtime.h>
#include <hip/hip_bf16.h>

typedef __bf16 bf16_t;
typedef __bf16 v16bf __attribute__((ext_vector_type(16)));
typedef float  v8f   __attribute__((ext_vector_type(8)));
typedef unsigned int v4u32 __attribute__((ext_vector_type(4)));
typedef int          v4i32 __attribute__((ext_vector_type(4)));
typedef int          v8i32 __attribute__((ext_vector_type(8)));

struct alignas(16) V4u { unsigned int x, y, z, w; };
union FragA { V4u q[2]; v16bf v; };

#define NWG      32
#define TPB      256
#define BATCH    64
#define TSTEPS   512
#define IN_DIM   512
#define HID      1024
#define KTPC     16      // kTiles (K=32) per 512-wide K chunk
#define NCHUNKS  3       // 1536 / 512

// workspace byte offsets
#define WS_XBF    0u              // 64*512*512 bf16   = 32 MB
#define WS_WPACK  33554432u       // 1536*4096 bf16    = 12 MB, fragment-packed
#define WS_HBF    46137344u       // 64*1024 bf16      = 128 KB
#define WS_CBUF   46268416u       // 64*1024 f32       = 256 KB
#define WS_BAR    46530560u       // 2 x u32 barrier

#define OUT_HSEQ  0u
#define OUT_HLAST 33554432u
#define OUT_CLAST 33619968u

// dynamic LDS layout (dynamic base == 0: no static LDS in this kernel)
#define SMEM_A0   0           // A chunk buffer 0: 64 x 512 bf16 = 65536 B
#define SMEM_A1   65536       // A chunk buffer 1: 65536 B
#define SMEM_C    131072      // [4][64][32] f32 = 32768 B
#define SMEM_B    163840      // [4][32] f32 bias = 512 B
#define SMEM_SZ   164352

#if __has_builtin(__builtin_amdgcn_tensor_load_to_lds)
#define HAVE_TDM 1
#else
#define HAVE_TDM 0
#endif

// ---------------- pre-pass kernels ----------------

__global__ void convert_x(const float* __restrict__ x, bf16_t* __restrict__ xb) {
  size_t i = ((size_t)blockIdx.x * TPB + threadIdx.x) * 8;
  alignas(16) bf16_t o[8];
#pragma unroll
  for (int r = 0; r < 8; ++r) o[r] = (bf16_t)x[i + r];
  *(V4u*)(xb + i) = *(const V4u*)o;
}

// Pack W (concat of W_i/W_f/W_g/W_o along columns) into per-fragment, per-lane
// contiguous layout for the v_wmma_f32_16x16x32_bf16 B-operand (k order shared with the A loads):
// lanes 0-15 -> N=lane, K 0..15 ; lanes 16-31 -> N=lane-16, K 16..31.
// Fragment order: [wg 0..31][nLocal 0..7][kTile 0..47][lane][16 bf16].
__global__ void pack_w(const float* __restrict__ W_i, const float* __restrict__ W_f,
                       const float* __restrict__ W_g, const float* __restrict__ W_o,
                       bf16_t* __restrict__ Wpack) {
  int gid   = blockIdx.x * TPB + threadIdx.x;   // 0 .. 393215
  int lane  = gid & 31;
  int rest  = gid >> 5;
  int kt    = rest % 48;
  int rest2 = rest / 48;
  int nl    = rest2 & 7;
  int wg    = rest2 >> 3;
  int g     = nl >> 1;
  int hcol  = wg * 32 + (nl & 1) * 16 + (lane & 15);
  int kbase = kt * 32 + (lane >> 4) * 16;
  const float* src = (g == 0) ? W_i : (g == 1) ? W_f : (g == 2) ? W_g : W_o;
  alignas(16) bf16_t tmp[16];
#pragma unroll
  for (int r = 0; r < 16; ++r)
    tmp[r] = (bf16_t)src[(size_t)(kbase + r) * HID + hcol];
  V4u* dst = (V4u*)(Wpack + (size_t)gid * 16);
  dst[0] = *(const V4u*)(tmp);
  dst[1] = *(const V4u*)(tmp + 8);
}

__global__ void init_state(bf16_t* __restrict__ hbf, float* __restrict__ cbuf,
                           unsigned* __restrict__ bar) {
  int i = blockIdx.x * TPB + threadIdx.x;
#pragma unroll
  for (int r = 0; r < 4; ++r) {
    int idx = i * 4 + r;
    hbf[idx]  = (bf16_t)0.0f;
    cbuf[idx] = 0.0f;
  }
  if (blockIdx.x == 0 && threadIdx.x < 2) bar[threadIdx.x] = 0;
}

// ---------------- TDM: async 2D tile load global -> LDS ----------------
// Builds a Tensor DMA Descriptor (D#) per CDNA5 ISA §8:
//   group0: count=1 | lds_addr | global_addr[56:0] | type=2
//   group1: data_size=2B, tensor_dim0/1, tile_dim0/1, tensor_dim0_stride
// Tile: tile_d1 rows x tile_d0 bf16 elements, row stride = stride0 elements.
__device__ __forceinline__ void tdm_load_2d(unsigned lds_off, const void* gptr,
                                            unsigned tensor_d0, unsigned tensor_d1,
                                            unsigned tile_d0, unsigned tile_d1,
                                            unsigned long long stride0) {
#if HAVE_TDM
  unsigned long long ga = (unsigned long long)(size_t)gptr;
  v4u32 g0;
  g0[0] = 1u;                                            // count=1, user mode
  g0[1] = lds_off;                                       // lds_addr (bytes)
  g0[2] = (unsigned)(ga & 0xFFFFFFFFu);                  // global_addr[31:0]
  g0[3] = (unsigned)((ga >> 32) & 0x1FFFFFFu) | (2u << 30); // addr[56:32] | type=2
  v8i32 g1;
  g1[0] = (int)(1u << 16);                               // wg_mask=0, data_size=2B
  g1[1] = (int)((tensor_d0 & 0xFFFFu) << 16);            // tensor_dim0[15:0]
  g1[2] = (int)((tensor_d0 >> 16) | ((tensor_d1 & 0xFFFFu) << 16));
  g1[3] = (int)((tensor_d1 >> 16) | (tile_d0 << 16));    // tile_dim0
  g1[4] = (int)(tile_d1 & 0xFFFFu);                      // tile_dim1 (tile_dim2=0)
  g1[5] = (int)(stride0 & 0xFFFFFFFFull);                // tensor_dim0_stride[31:0]
  g1[6] = (int)((stride0 >> 32) & 0xFFFFull);            // stride[47:32]
  g1[7] = 0;
  v4i32 z4 = {0, 0, 0, 0};
#if __clang_major__ >= 23
  v8i32 z8 = {0, 0, 0, 0, 0, 0, 0, 0};
  __builtin_amdgcn_tensor_load_to_lds(g0, g1, z4, z4, z8, 0);
#else
  __builtin_amdgcn_tensor_load_to_lds(g0, g1, z4, z4, 0);
#endif
#else
  (void)lds_off; (void)gptr; (void)tensor_d0; (void)tensor_d1;
  (void)tile_d0; (void)tile_d1; (void)stride0;
#endif
}

// ---------------- persistent LSTM kernel ----------------

__device__ __forceinline__ void grid_barrier(unsigned* cnt, unsigned* gen) {
  __syncthreads();
  if (threadIdx.x == 0) {
    __threadfence();
    unsigned g = __hip_atomic_load(gen, __ATOMIC_ACQUIRE, __HIP_MEMORY_SCOPE_AGENT);
    unsigned prev = __hip_atomic_fetch_add(cnt, 1u, __ATOMIC_ACQ_REL, __HIP_MEMORY_SCOPE_AGENT);
    if (prev == NWG - 1) {
      __hip_atomic_store(cnt, 0u, __ATOMIC_RELAXED, __HIP_MEMORY_SCOPE_AGENT);
      __hip_atomic_fetch_add(gen, 1u, __ATOMIC_ACQ_REL, __HIP_MEMORY_SCOPE_AGENT);
    } else {
      while (__hip_atomic_load(gen, __ATOMIC_ACQUIRE, __HIP_MEMORY_SCOPE_AGENT) == g)
        __builtin_amdgcn_s_sleep(2);
    }
  }
  __syncthreads();
}

__device__ __forceinline__ float sigm_(float x) { return 1.0f / (1.0f + __expf(-x)); }
__device__ __forceinline__ float tanh_(float x) {
  float e = __expf(-2.0f * x);
  return (1.0f - e) / (1.0f + e);
}

__global__ void lstm_kernel(const bf16_t* __restrict__ Xbf,
                            const bf16_t* __restrict__ Wpack,
                            bf16_t* __restrict__ hbf,
                            float* __restrict__ cbuf,
                            const float* __restrict__ b_i, const float* __restrict__ b_f,
                            const float* __restrict__ b_g, const float* __restrict__ b_o,
                            float* __restrict__ out,
                            unsigned* __restrict__ bar) {
  extern __shared__ __align__(16) char smem[];
  char*  Ab[2] = { smem + SMEM_A0, smem + SMEM_A1 };  // double-buffered A chunks
  float* Csh = (float*)(smem + SMEM_C);               // [gate][m][j]
  float* bsh = (float*)(smem + SMEM_B);               // [gate][j]

  const int tid  = threadIdx.x;
  const int lane = tid & 31;
  const int wv   = tid >> 5;              // 8 waves
  const int wg   = blockIdx.x;            // H-slice [wg*32, wg*32+32)
  const int half = lane >> 4;
  const int mrow = lane & 15;

  if (tid < 128) {
    int g = tid >> 5, j = tid & 31;
    const float* bp = (g == 0) ? b_i : (g == 1) ? b_f : (g == 2) ? b_g : b_o;
    bsh[tid] = bp[wg * 32 + j];
  }

  const bf16_t* wpw = Wpack + (size_t)(wg * 8 + wv) * 48 * 512;
  unsigned* cnt = bar;
  unsigned* gen = bar + 1;
  const int gate = wv >> 1;
  const int jb   = (wv & 1) * 16 + (lane & 15);

  for (int t = 0; t < TSTEPS; ++t) {
    v8f acc0 = {}, acc1 = {}, acc2 = {}, acc3 = {};

#if HAVE_TDM
    // kick off chunk 0 (x_t): 64 rows x 512 bf16, row stride = T*IN_DIM elems
    if (wv == 0)
      tdm_load_2d(SMEM_A0, Xbf + (size_t)t * IN_DIM,
                  IN_DIM, BATCH, IN_DIM, BATCH, (unsigned long long)(TSTEPS * IN_DIM));
#endif

    for (int ch = 0; ch < NCHUNKS; ++ch) {
#if HAVE_TDM
      if (wv == 0) {
        if (ch + 1 < NCHUNKS) {
          // prefetch next chunk (h halves): 64 rows x 512 bf16, row stride HID
          tdm_load_2d((ch + 1) & 1 ? SMEM_A1 : SMEM_A0,
                      hbf + (size_t)ch * 512,           // ch+1 -> h offset (ch)*512
                      IN_DIM, BATCH, IN_DIM, BATCH, (unsigned long long)HID);
          __builtin_amdgcn_s_wait_tensorcnt(1);  // oldest (chunk ch) complete
        } else {
          __builtin_amdgcn_s_wait_tensorcnt(0);  // last chunk complete
        }
      }
      __syncthreads();
#else
      // fallback: manual staging of chunk ch into Ab[ch&1]
      for (int u = tid; u < 4096; u += TPB) {
        int m = u >> 6, uo = u & 63;
        const V4u* src = (ch == 0)
            ? (const V4u*)(Xbf + (size_t)m * (TSTEPS * IN_DIM) + (size_t)t * IN_DIM + uo * 8)
            : (const V4u*)(hbf + (size_t)m * HID + (size_t)(ch - 1) * 512 + uo * 8);
        *(V4u*)(Ab[ch & 1] + (size_t)u * 16) = *src;
      }
      __syncthreads();
#endif
      const char* As = Ab[ch & 1];

#pragma unroll 4
      for (int kc = 0; kc < KTPC; ++kc) {
        int kt = ch * KTPC + kc;
        FragA bfrag;
        const V4u* bp = (const V4u*)(wpw + (size_t)kt * 512 + lane * 16);
        bfrag.q[0] = bp[0]; bfrag.q[1] = bp[1];
        const char* abase = As + (size_t)(kc * 32 + half * 16) * 2;
        FragA a0, a1, a2, a3;
        const char* p0 = abase + (size_t)(0 * 16 + mrow) * 1024;
        const char* p1 = abase + (size_t)(1 * 16 + mrow) * 1024;
        const char* p2 = abase + (size_t)(2 * 16 + mrow) * 1024;
        const char* p3 = abase + (size_t)(3 * 16 + mrow) * 1024;
        a0.q[0] = *(const V4u*)p0;        a0.q[1] = *(const V4u*)(p0 + 16);
        a1.q[0] = *(const V4u*)p1;        a1.q[1] = *(const V4u*)(p1 + 16);
        a2.q[0] = *(const V4u*)p2;        a2.q[1] = *(const V4u*)(p2 + 16);
        a3.q[0] = *(const V4u*)p3;        a3.q[1] = *(const V4u*)(p3 + 16);
        acc0 = __builtin_amdgcn_wmma_f32_16x16x32_bf16(false, a0.v, false, bfrag.v, (short)0, acc0, false, false);
        acc1 = __builtin_amdgcn_wmma_f32_16x16x32_bf16(false, a1.v, false, bfrag.v, (short)0, acc1, false, false);
        acc2 = __builtin_amdgcn_wmma_f32_16x16x32_bf16(false, a2.v, false, bfrag.v, (short)0, acc2, false, false);
        acc3 = __builtin_amdgcn_wmma_f32_16x16x32_bf16(false, a3.v, false, bfrag.v, (short)0, acc3, false, false);
      }
      __syncthreads();
    }

    // prefetch next timestep's x rows into GL2 while we finish this step
    if (t + 1 < TSTEPS && tid < BATCH)
      __builtin_prefetch((const void*)(Xbf + (size_t)tid * (TSTEPS * IN_DIM) +
                                       (size_t)(t + 1) * IN_DIM), 0, 1);

    // spill accumulators (pre-activation gates) to LDS
#pragma unroll
    for (int r = 0; r < 8; ++r) {
      Csh[((gate * 64 + (0 * 16 + r + half * 8)) << 5) + jb] = acc0[r];
      Csh[((gate * 64 + (1 * 16 + r + half * 8)) << 5) + jb] = acc1[r];
      Csh[((gate * 64 + (2 * 16 + r + half * 8)) << 5) + jb] = acc2[r];
      Csh[((gate * 64 + (3 * 16 + r + half * 8)) << 5) + jb] = acc3[r];
    }

    grid_barrier(cnt, gen);   // all WGs finished reading hbf for step t

    // elementwise cell update for this WG's 32 hidden columns, all 64 batches
#pragma unroll
    for (int i = 0; i < 8; ++i) {
      int e = tid * 8 + i;
      int m = e >> 5, j = e & 31;
      int hcol = wg * 32 + j;
      float xi = Csh[(((0 * 64) + m) << 5) + j] + bsh[j];
      float xf = Csh[(((1 * 64) + m) << 5) + j] + bsh[32 + j];
      float xg = Csh[(((2 * 64) + m) << 5) + j] + bsh[64 + j];
      float xo = Csh[(((3 * 64) + m) << 5) + j] + bsh[96 + j];
      float it = sigm_(xi), ft = sigm_(xf), gt = tanh_(xg), ot = sigm_(xo);
      size_t ci = (size_t)m * HID + hcol;
      float cn = ft * cbuf[ci] + it * gt;
      float hn = ot * tanh_(cn);
      cbuf[ci] = cn;
      hbf[ci]  = (bf16_t)hn;
      out[OUT_HSEQ + (size_t)m * (TSTEPS * HID) + (size_t)t * HID + hcol] = hn;
      if (t == TSTEPS - 1) {
        out[OUT_HLAST + ci] = hn;
        out[OUT_CLAST + ci] = cn;
      }
    }

    grid_barrier(cnt, gen);   // h_{t+1} inputs visible to all WGs
  }
}

// ---------------- host launcher ----------------

extern "C" void kernel_launch(void* const* d_in, const int* in_sizes, int n_in,
                              void* d_out, int out_size, void* d_ws, size_t ws_size,
                              hipStream_t stream) {
  const float* x   = (const float*)d_in[0];
  const float* W_i = (const float*)d_in[1];
  const float* b_i = (const float*)d_in[2];
  const float* W_f = (const float*)d_in[3];
  const float* b_f = (const float*)d_in[4];
  const float* W_g = (const float*)d_in[5];
  const float* b_g = (const float*)d_in[6];
  const float* W_o = (const float*)d_in[7];
  const float* b_o = (const float*)d_in[8];
  float* out = (float*)d_out;
  char*  ws  = (char*)d_ws;

  bf16_t*   Xbf   = (bf16_t*)(ws + WS_XBF);
  bf16_t*   Wpack = (bf16_t*)(ws + WS_WPACK);
  bf16_t*   hbf   = (bf16_t*)(ws + WS_HBF);
  float*    cbuf  = (float*)(ws + WS_CBUF);
  unsigned* bar   = (unsigned*)(ws + WS_BAR);

  convert_x<<<8192, TPB, 0, stream>>>(x, Xbf);                       // 16.7M elems
  pack_w<<<1536, TPB, 0, stream>>>(W_i, W_f, W_g, W_o, Wpack);       // 393216 lane-frags
  init_state<<<64, TPB, 0, stream>>>(hbf, cbuf, bar);
  lstm_kernel<<<NWG, TPB, SMEM_SZ, stream>>>(Xbf, Wpack, hbf, cbuf,
                                             b_i, b_f, b_g, b_o, out, bar);
}
